// HaloAttention_1236950581792
// MI455X (gfx1250) — compile-verified, matrix operand order checked by reading, not run on hardware
//
#include <hip/hip_runtime.h>

#define HEADS 8
#define BSZ   8
#define HALO  3
#define WIN   14          // BSZ + 2*HALO
#define NP    196         // WIN*WIN real kv positions
#define NPT   208         // padded to 13 tiles of 16
#define NKT   13          // n-tiles over kv positions
#define CIN   256
#define DH    32          // dhqk == dhv
#define XW_LD 264         // x-window row stride (bf16 elems), padded vs 256 to rotate LDS banks
#define PS_LD 224         // P-scratch row stride: 7 K-tiles of 32

typedef __attribute__((ext_vector_type(16))) __bf16 bf16x16;
typedef __attribute__((ext_vector_type(8)))  float  f32x8;

// A/B fragment loader: e[0..7] at p, e[8..15] at p+16 elements (both 16B, 16B-aligned)
__device__ __forceinline__ bf16x16 ld_frag(const __bf16* p) {
  union { bf16x16 v; uint4 q[2]; } u;
  u.q[0] = *(const uint4*)(p);
  u.q[1] = *(const uint4*)(p + 16);
  return u.v;
}

__device__ __forceinline__ f32x8 wmma_bf16(bf16x16 a, bf16x16 b, f32x8 c) {
  return __builtin_amdgcn_wmma_f32_16x16x32_bf16(false, a, false, b, (short)0, c,
                                                 false, false);
}

// ---- pre-pass: fp32 weights -> bf16 in workspace (L2-resident, reused 1024x) ----
__global__ void cvt_w(const float* __restrict__ Wq, const float* __restrict__ Wkv,
                      __bf16* __restrict__ wq_bf, __bf16* __restrict__ wkv_bf) {
  int i = blockIdx.x * 256 + threadIdx.x;
  if (i < 256 * 256) wq_bf[i]  = (__bf16)Wq[i];
  if (i < 512 * 256) wkv_bf[i] = (__bf16)Wkv[i];
}

__global__ void __launch_bounds__(256)
halo_attn(const float* __restrict__ x, const float* __restrict__ pos_bias,
          const __bf16* __restrict__ wq_bf, const __bf16* __restrict__ wkv_bf,
          float* __restrict__ out) {
  extern __shared__ __bf16 smem[];
  __bf16* xw = smem;                          // [NPT][XW_LD]  107KB (region A)
  __bf16* kl = smem + NPT * XW_LD;            // [HEADS][NPT][DH]  104KB
  __bf16* vT = kl + HEADS * NPT * DH;         // [HEADS][DH][NPT]  104KB (transposed V)
  __bf16* qs = smem;                          // [HEADS][64][DH]   aliases region A
  __bf16* ps = smem + HEADS * 64 * DH;        // [HEADS][16][PS_LD] aliases region A

  const int tid  = threadIdx.x;
  const int lane = tid & 31;
  const int wv   = tid >> 5;                  // wave id == head id
  const int l16  = lane & 15;
  const int hi   = (lane >> 4) & 1;

  const int bid = blockIdx.x;
  const int b   = bid >> 6;
  const int blk = bid & 63;
  const int bh  = blk >> 3;
  const int bw  = blk & 7;
  const int h0  = bh * BSZ - HALO;
  const int w0  = bw * BSZ - HALO;

  // ---------- Stage 1: x window -> LDS bf16, zero-padded (halo OOB + pad rows) ----------
  if (tid < NPT) {
    const int p  = tid;
    const int hh = h0 + p / WIN;
    const int ww = w0 + p % WIN;
    const bool in = (p < NP) && (hh >= 0) && (hh < 64) && (ww >= 0) && (ww < 64);
    __bf16* dst = xw + p * XW_LD;
    if (in) {
      const float* xp = x + (size_t)b * CIN * 4096 + hh * 64 + ww;
      for (int c = 0; c < CIN; ++c) dst[c] = (__bf16)xp[(size_t)c * 4096];
    } else {
      for (int c = 0; c < CIN; ++c) dst[c] = (__bf16)0.0f;
    }
  }
  __syncthreads();

  // ---------- Stage 2: kv projection [NPT x 256] @ [256 x 512] ----------
  // nt-outer: keep this wave's weight B-fragments in registers across all 13 mt tiles.
#pragma unroll
  for (int nn = 0; nn < 4; ++nn) {
    const int nt = wv + nn * HEADS;           // this wave's 4 output-column tiles
    const int o  = nt * 16 + l16;
    const __bf16* brow = wkv_bf + (size_t)o * CIN + hi * 8;
    bf16x16 bfr[8];
#pragma unroll
    for (int kt = 0; kt < 8; ++kt) bfr[kt] = ld_frag(brow + kt * 32);

    const int head = nt >> 2;                 // (nt*16)/64
    const int d0   = (nt & 3) * 16;           // (nt*16)%64
    for (int mt = 0; mt < NKT; ++mt) {
      const __bf16* arow = xw + (mt * 16 + l16) * XW_LD + hi * 8;
      f32x8 acc = {};
#pragma unroll
      for (int kt = 0; kt < 8; ++kt)
        acc = wmma_bf16(ld_frag(arow + kt * 32), bfr[kt], acc);

      if (d0 < DH) {                          // K tile -> row-major [p][d]
        const int d = d0 + l16;
#pragma unroll
        for (int r = 0; r < 8; ++r) {
          const int pp = mt * 16 + hi * 8 + r;
          kl[(head * NPT + pp) * DH + d] = (__bf16)acc[r];
        }
      } else {                                // V tile -> transposed [d][p], packed b128
        const int d = d0 - DH + l16;
        union { __bf16 h8[8]; uint4 q; } pk;
#pragma unroll
        for (int r = 0; r < 8; ++r) pk.h8[r] = (__bf16)acc[r];
        *(uint4*)(vT + (head * DH + d) * NPT + mt * 16 + hi * 8) = pk.q;
      }
    }
  }

  // ---------- Stage 3: q projection for this wave's head (64x32), B hoisted ----------
  f32x8 qacc[4][2];
#pragma unroll
  for (int nt = 0; nt < 2; ++nt) {
    const int o = wv * DH + nt * 16 + l16;
    const __bf16* brow = wq_bf + (size_t)o * CIN + hi * 8;
    bf16x16 bfr[8];
#pragma unroll
    for (int kt = 0; kt < 8; ++kt) bfr[kt] = ld_frag(brow + kt * 32);
#pragma unroll
    for (int mt = 0; mt < 4; ++mt) {
      const int m = mt * 16 + l16;
      const int p = (HALO + (m >> 3)) * WIN + HALO + (m & 7);  // central 8x8
      const __bf16* arow = xw + p * XW_LD + hi * 8;
      f32x8 acc = {};
#pragma unroll
      for (int kt = 0; kt < 8; ++kt)
        acc = wmma_bf16(ld_frag(arow + kt * 32), bfr[kt], acc);
      qacc[mt][nt] = acc;
    }
  }
  __syncthreads();   // xw dead: region A reusable; kv fully written

  // ---------- Stage 4: Q (D-layout) -> LDS (A-layout source); zero P pad cols ----------
#pragma unroll
  for (int mt = 0; mt < 4; ++mt)
#pragma unroll
    for (int nt = 0; nt < 2; ++nt)
#pragma unroll
      for (int r = 0; r < 8; ++r) {
        const int qp = mt * 16 + hi * 8 + r;
        qs[(wv * 64 + qp) * DH + nt * 16 + l16] = (__bf16)qacc[mt][nt][r];
      }
  {
    __bf16* psw = ps + wv * 16 * PS_LD;
#pragma unroll
    for (int r = 0; r < 8; ++r)
      psw[(hi * 8 + r) * PS_LD + NPT + l16] = (__bf16)0.0f;
  }
  __syncthreads();

  bf16x16 aq[4];
#pragma unroll
  for (int mt = 0; mt < 4; ++mt)
    aq[mt] = ld_frag(qs + (wv * 64 + mt * 16 + l16) * DH + hi * 8);

  // ---------- Stage 5: attention, wave wv == head wv ----------
  const float scale = 0.17677669529663687f;   // 32^-0.5
  const float* pb = pos_bias + (size_t)wv * 64 * NP;
  __bf16* psw = ps + wv * 16 * PS_LD;

  for (int mt = 0; mt < 4; ++mt) {
    f32x8 s[NKT];
#pragma unroll
    for (int nt = 0; nt < NKT; ++nt) {        // S = Q @ K^T  (K=32, one WMMA/tile)
      const __bf16* kb = kl + (wv * NPT + nt * 16 + l16) * DH + hi * 8;
      f32x8 z = {};
      s[nt] = wmma_bf16(aq[mt], ld_frag(kb), z);
    }

    float rmax[8], rsum[8];
#pragma unroll
    for (int r = 0; r < 8; ++r) rmax[r] = -3.0e38f;
#pragma unroll
    for (int nt = 0; nt < NKT; ++nt) {
      const int kp = nt * 16 + l16;
#pragma unroll
      for (int r = 0; r < 8; ++r) {
        const int qp = mt * 16 + hi * 8 + r;
        float v = (kp < NP) ? (s[nt][r] * scale + pb[qp * NP + kp]) : -1.0e30f;
        s[nt][r] = v;
        rmax[r]  = fmaxf(rmax[r], v);
      }
    }
#pragma unroll
    for (int r = 0; r < 8; ++r)
      for (int off = 1; off < 16; off <<= 1)
        rmax[r] = fmaxf(rmax[r], __shfl_xor(rmax[r], off, 32));
#pragma unroll
    for (int r = 0; r < 8; ++r) rsum[r] = 0.0f;
#pragma unroll
    for (int nt = 0; nt < NKT; ++nt)
#pragma unroll
      for (int r = 0; r < 8; ++r) {
        float e = __expf(s[nt][r] - rmax[r]);
        s[nt][r] = e;
        rsum[r] += e;
      }
#pragma unroll
    for (int r = 0; r < 8; ++r)
      for (int off = 1; off < 16; off <<= 1)
        rsum[r] += __shfl_xor(rsum[r], off, 32);

    // P (bf16) -> per-wave LDS scratch, then consume as A-fragments
#pragma unroll
    for (int nt = 0; nt < NKT; ++nt)
#pragma unroll
      for (int r = 0; r < 8; ++r)
        psw[(hi * 8 + r) * PS_LD + nt * 16 + l16] = (__bf16)s[nt][r];
    asm volatile("s_wait_dscnt 0" ::: "memory");  // same-wave LDS RAW

    f32x8 oacc[2] = {};
#pragma unroll
    for (int kt = 0; kt < 7; ++kt) {          // O = P @ V, K = 224 (pad cols are 0)
      bf16x16 a = ld_frag(psw + l16 * PS_LD + kt * 32 + hi * 8);
#pragma unroll
      for (int nt = 0; nt < 2; ++nt) {
        const __bf16* vb = vT + (wv * DH + nt * 16 + l16) * NPT + kt * 32 + hi * 8;
        oacc[nt] = wmma_bf16(a, ld_frag(vb), oacc[nt]);
      }
    }

    float rinv[8];
#pragma unroll
    for (int r = 0; r < 8; ++r) rinv[r] = 1.0f / rsum[r];

    // lane's 8 rows are 8 consecutive w positions -> two 16B vector stores per nt
    const int qp0 = mt * 16 + hi * 8;         // multiple of 8 -> fixed output row
    const int qr  = qp0 >> 3;
#pragma unroll
    for (int nt = 0; nt < 2; ++nt) {
      const int ch = wv * DH + nt * 16 + l16;
      float* op = out + (((size_t)b * 256 + ch) * 64 + bh * BSZ + qr) * 64 + bw * BSZ;
      float4 v0, v1;
      v0.x = oacc[nt][0] * rinv[0];  v0.y = oacc[nt][1] * rinv[1];
      v0.z = oacc[nt][2] * rinv[2];  v0.w = oacc[nt][3] * rinv[3];
      v1.x = oacc[nt][4] * rinv[4];  v1.y = oacc[nt][5] * rinv[5];
      v1.z = oacc[nt][6] * rinv[6];  v1.w = oacc[nt][7] * rinv[7];
      *(float4*)(op)     = v0;
      *(float4*)(op + 4) = v1;
    }
  }
}

extern "C" void kernel_launch(void* const* d_in, const int* in_sizes, int n_in,
                              void* d_out, int out_size, void* d_ws, size_t ws_size,
                              hipStream_t stream) {
  (void)in_sizes; (void)n_in; (void)out_size; (void)ws_size;
  const float* x   = (const float*)d_in[0];
  const float* Wq  = (const float*)d_in[1];
  const float* Wkv = (const float*)d_in[2];
  const float* pb  = (const float*)d_in[3];
  float* out = (float*)d_out;

  __bf16* wq_bf  = (__bf16*)d_ws;            // 256*256 bf16
  __bf16* wkv_bf = wq_bf + 256 * 256;        // 512*256 bf16  (total 384KB of ws)

  cvt_w<<<dim3(512), dim3(256), 0, stream>>>(Wq, Wkv, wq_bf, wkv_bf);

  const size_t lds_bytes =
      (size_t)(NPT * XW_LD + HEADS * NPT * DH + HEADS * DH * NPT) * sizeof(__bf16); // 322,816B
  hipFuncSetAttribute((const void*)halo_attn,
                      hipFuncAttributeMaxDynamicSharedMemorySize, (int)lds_bytes);
  halo_attn<<<dim3(16 * 64), dim3(256), lds_bytes, stream>>>(x, pb, wq_bf, wkv_bf, out);
}
